// MultiHeadAttention_18245021073929
// MI455X (gfx1250) — matrix operand order repaired and checked by reading, not compiled, not run on hardware
//
#include <hip/hip_runtime.h>
#include <hip/hip_bf16.h>
#include <math.h>

typedef __attribute__((ext_vector_type(16))) _Float16 v16h;
typedef __attribute__((ext_vector_type(8)))  _Float16 v8h;
typedef __attribute__((ext_vector_type(4)))  _Float16 v4h;
typedef __attribute__((ext_vector_type(8)))  float    v8f;

#define BATCH 4
#define SEQ   2048
#define HID   1024
#define NHEAD 16
#define HDIM  64

// ---------------------------------------------------------------------------
// CDNA5 async global->LDS copy (16B per lane), tracked by ASYNCcnt.
// GV mode: LDS address comes from the vdst VGPR, global address from the 64-bit vaddr.
// ---------------------------------------------------------------------------
#define ASYNC_CP16(ldsoff, gsrc)                                        \
  asm volatile("global_load_async_to_lds_b128 %0, %1, off"              \
               :: "v"(ldsoff), "v"(gsrc) : "memory")
#define WAIT_ASYNC0() asm volatile("s_wait_asynccnt 0x0" ::: "memory")
#define WAIT_ASYNC4() asm volatile("s_wait_asynccnt 0x4" ::: "memory")

__device__ __forceinline__ unsigned lds_off(const void* p) {
  // addrspacecast(LDS->flat) puts the LDS byte offset in the low 32 bits
  return (unsigned)(uintptr_t)p;
}

// ---------------------------------------------------------------------------
// WMMA fragment helpers (v_wmma_f32_16x16x32_f16, wave32).
// A 16x32 (MxK) f16: lane l -> row l%16, g=l/16; k chunks [8g..8g+7], [16+8g..]
// B 32x16 (KxN) f16 supplied row-major [N x K] ("NT"): lane->col l%16, k chunks [8g..8g+7], [16+8g..]
// C/D 16x16 f32: VGPR r -> row r + 8g, col l%16
// ---------------------------------------------------------------------------
__device__ __forceinline__ v8f vzero8() {
  v8f z;
#pragma unroll
  for (int i = 0; i < 8; ++i) z[i] = 0.0f;
  return z;
}

__device__ __forceinline__ v16h load_a_frag(const _Float16* base, int ld, int lane) {
  const int m = lane & 15;
  const int g = lane >> 4;
  const _Float16* p = base + (size_t)m * ld + 8 * g;
  v8h lo = *(const v8h*)(p);        // k = 8g .. 8g+7
  v8h hi = *(const v8h*)(p + 16);   // k = 16+8g .. 16+8g+7
  v16h r;
#pragma unroll
  for (int i = 0; i < 8; ++i) { r[i] = lo[i]; r[i + 8] = hi[i]; }
  return r;
}

__device__ __forceinline__ v16h load_b_frag(const _Float16* base, int ld, int lane) {
  const int n = lane & 15;
  const int g = lane >> 4;
  const _Float16* p = base + (size_t)n * ld + 8 * g;
  v8h lo = *(const v8h*)(p);
  v8h hi = *(const v8h*)(p + 16);
  v16h r;
#pragma unroll
  for (int i = 0; i < 8; ++i) { r[i] = lo[i]; r[i + 8] = hi[i]; }
  return r;
}

__device__ __forceinline__ v8f wmma_f16(v16h a, v16h b, v8f c) {
  return __builtin_amdgcn_wmma_f32_16x16x32_f16(false, a, false, b, (short)0, c,
                                                false, false);
}

// ---------------------------------------------------------------------------
// f32 -> f16 bulk convert (x and the three weight matrices), 4 elems/thread
// ---------------------------------------------------------------------------
__global__ __launch_bounds__(256) void cvt_f16_kernel(
    const float* __restrict__ in, _Float16* __restrict__ out, int n4) {
  const int i = blockIdx.x * 256 + threadIdx.x;
  if (i < n4) {
    const float4 v = ((const float4*)in)[i];
    v4h o;
    o[0] = (_Float16)v.x; o[1] = (_Float16)v.y;
    o[2] = (_Float16)v.z; o[3] = (_Float16)v.w;
    ((v4h*)out)[i] = o;
  }
}

// ---------------------------------------------------------------------------
// QKV projection: Out = Xh * Wh^T + bias (all-f16 operands, f32 accum).
// Double-buffered async LDS tiles, 128(M) x 128(N) x 32(K), 8 waves.
// mode 0: store f16 [B, NH, S, 64] (Q,K); mode 1: [B, NH, 64, S] (V^T)
// ---------------------------------------------------------------------------
#define PJLD 40  // tile stride in halves (80B rows: 16B aligned, bank-spread)

__global__ __launch_bounds__(256) void qkv_proj_kernel(
    const _Float16* __restrict__ Xh, const _Float16* __restrict__ Wh,
    const float* __restrict__ bias, _Float16* __restrict__ Out, int mode) {
  __shared__ __align__(16) _Float16 As[2][128 * PJLD];
  __shared__ __align__(16) _Float16 Bs[2][128 * PJLD];

  const int tid  = threadIdx.x;
  const int lane = tid & 31;
  const int wv   = tid >> 5;
  const int g    = lane >> 4;
  const int ln   = lane & 15;
  const int nBase = blockIdx.x * 128;
  const int mBase = blockIdx.y * 128;

  auto issue = [&](int buf, int k0) {
#pragma unroll
    for (int i = 0; i < 2; ++i) {
      const int c = i * 256 + tid;           // 0..511 16B-chunks per tile
      const int row = c >> 2, off = (c & 3) * 8;
      ASYNC_CP16(lds_off(&As[buf][row * PJLD + off]),
                 Xh + (size_t)(mBase + row) * HID + k0 + off);
      ASYNC_CP16(lds_off(&Bs[buf][row * PJLD + off]),
                 Wh + (size_t)(nBase + row) * HID + k0 + off);
    }
  };

  v8f acc[8];
#pragma unroll
  for (int i = 0; i < 8; ++i) acc[i] = vzero8();

  issue(0, 0);
  for (int it = 0; it < HID / 32; ++it) {
    const int buf = it & 1;
    if (it + 1 < HID / 32) { issue(buf ^ 1, (it + 1) * 32); WAIT_ASYNC4(); }
    else                   { WAIT_ASYNC0(); }
    __syncthreads();

    const v16h a = load_a_frag(&As[buf][(wv * 16) * PJLD], PJLD, lane);
#pragma unroll
    for (int nt = 0; nt < 8; ++nt) {
      const v16h b = load_b_frag(&Bs[buf][(nt * 16) * PJLD], PJLD, lane);
      acc[nt] = wmma_f16(a, b, acc[nt]);
    }
    __syncthreads();
  }

#pragma unroll
  for (int nt = 0; nt < 8; ++nt) {
    const int n  = nBase + nt * 16 + ln;
    const float bvv = bias[n];
    const int h = n >> 6, d = n & 63;
#pragma unroll
    for (int r = 0; r < 8; ++r) {
      const int m = mBase + wv * 16 + r + 8 * g;
      const int b = m >> 11, s = m & 2047;
      const float val = acc[nt][r] + bvv;
      size_t idx;
      if (mode == 0) idx = (((size_t)b * NHEAD + h) * SEQ + s) * HDIM + d;
      else           idx = (((size_t)b * NHEAD + h) * HDIM + d) * SEQ + s;
      Out[idx] = (_Float16)val;
    }
  }
}

// ---------------------------------------------------------------------------
// Flash attention: block per (b*NH+h, 128-query block); 64-key tiles.
// K/V tiles double-buffered via async LDS; scores f16 in LDS, softmax
// rewrites probabilities in place (one thread owns one query row).
// ---------------------------------------------------------------------------
#define PLD 72  // SP stride (halves)
#define KLD 72  // K/V tile stride (halves)

__global__ __launch_bounds__(256) void attn_kernel(
    const _Float16* __restrict__ Qf, const _Float16* __restrict__ Kf,
    const _Float16* __restrict__ Vt, const float* __restrict__ mask,
    float* __restrict__ Out) {
  __shared__ __align__(16) _Float16 SP[128 * PLD];      // scores -> probs
  __shared__ __align__(16) _Float16 Ks[2][64 * KLD];
  __shared__ __align__(16) _Float16 Vs[2][64 * KLD];
  __shared__ float corr_s[128];
  __shared__ float rsum_s[128];

  const int tid  = threadIdx.x;
  const int lane = tid & 31;
  const int wv   = tid >> 5;
  const int g    = lane >> 4;
  const int ln   = lane & 15;
  const int bh   = blockIdx.y;     // 0..63
  const int b    = bh >> 4;
  const int qb   = blockIdx.x;     // 0..15

  const _Float16* Qbase = Qf + ((size_t)bh * SEQ + (size_t)qb * 128) * HDIM;
  const _Float16* Kbase = Kf + (size_t)bh * SEQ * HDIM;
  const _Float16* Vbase = Vt + (size_t)bh * HDIM * SEQ;   // [64][2048]
  const float*    mb    = mask + (size_t)b * SEQ;

  auto issue = [&](int buf, int kb) {
#pragma unroll
    for (int i = 0; i < 2; ++i) {
      const int c = i * 256 + tid;           // 0..511 16B-chunks per tile
      const int row = c >> 3, off = (c & 7) * 8;
      ASYNC_CP16(lds_off(&Ks[buf][row * KLD + off]),
                 Kbase + (size_t)(kb + row) * HDIM + off);
      ASYNC_CP16(lds_off(&Vs[buf][row * KLD + off]),
                 Vbase + (size_t)row * SEQ + kb + off);
    }
  };

  v16h qa[2];
#pragma unroll
  for (int kt = 0; kt < 2; ++kt)
    qa[kt] = load_a_frag(Qbase + (size_t)(wv * 16) * HDIM + kt * 32, HDIM, lane);

  v8f oacc[4];
#pragma unroll
  for (int i = 0; i < 4; ++i) oacc[i] = vzero8();

  float run_max = -INFINITY;  // thread tid (<128) owns query row tid
  float run_sum = 0.0f;

  issue(0, 0);
  for (int it = 0; it < SEQ / 64; ++it) {
    const int kb = it * 64;
    const int buf = it & 1;
    if (it + 1 < SEQ / 64) { issue(buf ^ 1, kb + 64); WAIT_ASYNC4(); }
    else                   { WAIT_ASYNC0(); }
    __syncthreads();

    // ---- step 1: scores = Q K^T (128x64 tile) -> SP (f16) ----
#pragma unroll
    for (int nt = 0; nt < 4; ++nt) {
      v8f c = vzero8();
#pragma unroll
      for (int kt = 0; kt < 2; ++kt) {
        const v16h bf = load_b_frag(&Ks[buf][(nt * 16) * KLD + kt * 32], KLD, lane);
        c = wmma_f16(qa[kt], bf, c);
      }
#pragma unroll
      for (int r = 0; r < 8; ++r)
        SP[(wv * 16 + r + 8 * g) * PLD + nt * 16 + ln] = (_Float16)c[r];
    }
    __syncthreads();

    // ---- step 2: online softmax, in-place score->prob rewrite ----
    if (tid < 128) {
      _Float16* srow = &SP[tid * PLD];
      float tmax = -INFINITY;
#pragma unroll 8
      for (int j = 0; j < 64; ++j) {
        const float s = (float)srow[j] * 0.125f + mb[kb + j];
        tmax = fmaxf(tmax, s);
      }
      const float nmax = fmaxf(run_max, tmax);
      const float corr = __expf(run_max - nmax);  // 0 on first tile
      float lsum = 0.0f;
#pragma unroll 8
      for (int j = 0; j < 64; ++j) {
        const float s = (float)srow[j] * 0.125f + mb[kb + j];
        const float p = __expf(s - nmax);
        lsum += p;
        srow[j] = (_Float16)p;   // same thread, read-then-write: race-free
      }
      run_sum = run_sum * corr + lsum;
      run_max = nmax;
      corr_s[tid] = corr;
    }
    __syncthreads();

    // ---- step 3: rescale O, then O += P * V ----
#pragma unroll
    for (int r = 0; r < 8; ++r) {
      const float cf = corr_s[wv * 16 + r + 8 * g];
#pragma unroll
      for (int nt = 0; nt < 4; ++nt) oacc[nt][r] *= cf;
    }
#pragma unroll
    for (int kt = 0; kt < 2; ++kt) {
      const v16h pa = load_a_frag(&SP[(wv * 16) * PLD + kt * 32], PLD, lane);
#pragma unroll
      for (int nt = 0; nt < 4; ++nt) {
        const v16h bf = load_b_frag(&Vs[buf][(nt * 16) * KLD + kt * 32], KLD, lane);
        oacc[nt] = wmma_f16(pa, bf, oacc[nt]);
      }
    }
    __syncthreads();
  }

  if (tid < 128) rsum_s[tid] = run_sum;
  __syncthreads();

  float* ob = Out + ((size_t)b * SEQ + (size_t)qb * 128) * HID + (bh & 15) * HDIM;
#pragma unroll
  for (int r = 0; r < 8; ++r) {
    const int m = wv * 16 + r + 8 * g;
    const float inv = 1.0f / rsum_s[m];
#pragma unroll
    for (int nt = 0; nt < 4; ++nt)
      ob[(size_t)m * HID + nt * 16 + ln] = oacc[nt][r] * inv;
  }
}

// ---------------------------------------------------------------------------
// Launch
// ---------------------------------------------------------------------------
extern "C" void kernel_launch(void* const* d_in, const int* in_sizes, int n_in,
                              void* d_out, int out_size, void* d_ws, size_t ws_size,
                              hipStream_t stream) {
  (void)in_sizes; (void)n_in; (void)out_size; (void)ws_size;
  const float* x    = (const float*)d_in[0];
  const float* amask= (const float*)d_in[1];
  const float* Wq   = (const float*)d_in[2];
  const float* bq   = (const float*)d_in[3];
  const float* Wk   = (const float*)d_in[4];
  const float* bk   = (const float*)d_in[5];
  const float* Wv   = (const float*)d_in[6];
  const float* bv   = (const float*)d_in[7];

  const size_t qkv = (size_t)BATCH * NHEAD * SEQ * HDIM;  // 8,388,608 halves
  const size_t wsz = (size_t)HID * HID;                   // 1,048,576 halves
  _Float16* Qh  = (_Float16*)d_ws;
  _Float16* Kh  = Qh + qkv;
  _Float16* Vh  = Kh + qkv;     // stored transposed per head: [B,NH,64,S]
  _Float16* Xh  = Vh + qkv;
  _Float16* Wqh = Xh + qkv;
  _Float16* Wkh = Wqh + wsz;
  _Float16* Wvh = Wkh + wsz;

  // bulk f32->f16 converts
  cvt_f16_kernel<<<(int)(qkv / 1024), 256, 0, stream>>>(x,  Xh,  (int)(qkv / 4));
  cvt_f16_kernel<<<(int)(wsz / 1024), 256, 0, stream>>>(Wq, Wqh, (int)(wsz / 4));
  cvt_f16_kernel<<<(int)(wsz / 1024), 256, 0, stream>>>(Wk, Wkh, (int)(wsz / 4));
  cvt_f16_kernel<<<(int)(wsz / 1024), 256, 0, stream>>>(Wv, Wvh, (int)(wsz / 4));

  dim3 pblk(256);
  dim3 pgrd(HID / 128, (BATCH * SEQ) / 128);  // (8, 64)
  qkv_proj_kernel<<<pgrd, pblk, 0, stream>>>(Xh, Wqh, bq, Qh, 0);
  qkv_proj_kernel<<<pgrd, pblk, 0, stream>>>(Xh, Wkh, bk, Kh, 0);
  qkv_proj_kernel<<<pgrd, pblk, 0, stream>>>(Xh, Wvh, bv, Vh, 1);

  dim3 agrd(SEQ / 128, BATCH * NHEAD);        // (16, 64)
  attn_kernel<<<agrd, pblk, 0, stream>>>(Qh, Kh, Vh, amask, (float*)d_out);
}
